// FuzzyCRF_48679159332974
// MI455X (gfx1250) — compile-verified, matrix operand order checked by reading, not run on hardware
//
#include <hip/hip_runtime.h>

// FuzzyCRF forward/gold scan for MI455X (gfx1250, wave32, WMMA).
// B=64, S=1024, T=128. Latency-bound sequential scan; per-step logsumexp
// recast as exp-space GEMM so v_wmma_f32_16x16x32_f16 does the T x T reduction.
// This revision reduces the per-step barrier count from 4 to 2 by folding the
// cross-wave row-max reduction into the WMMA/update phase via 16-lane
// shfl-xor butterflies (wave-local, no workgroup sync).

#define BB   64
#define SS   1024
#define TT   128
#define NEGV (-1.0e7f)

typedef _Float16 v8h  __attribute__((ext_vector_type(8)));
typedef _Float16 v16h __attribute__((ext_vector_type(16)));
typedef float    v8f  __attribute__((ext_vector_type(8)));

static __device__ __forceinline__ v16h cat8(v8h lo, v8h hi) {
  return __builtin_shufflevector(lo, hi, 0,1,2,3,4,5,6,7,8,9,10,11,12,13,14,15);
}

__global__ __launch_bounds__(256) void crf_scan_kernel(
    const float* __restrict__ feats,     // (B,S,T) f32
    const int*   __restrict__ maskI,     // (B,S) i32
    const int*   __restrict__ ptags,     // (B,S,T) i32
    const float* __restrict__ trans,     // (T,T) f32
    const float* __restrict__ startt,    // (T,) f32
    const float* __restrict__ endt,      // (T,) f32
    float*       __restrict__ partial)   // (4,) f32 workspace
{
  // ---- LDS-resident scan state (never touches HBM inside the loop) ----
  __shared__ float alphaF[16][TT];                 // forward alpha, 8 KB
  __shared__ float alphaG[16][TT];                 // gold alpha,    8 KB
  __shared__ alignas(16) _Float16 vhF[16][TT];     // exp(alpha-m) f16, 4 KB
  __shared__ alignas(16) _Float16 vhG[16][TT];     // masked gold version, 4 KB
  __shared__ float tileMaxF[16][8];                // per-(row, wave) max, fwd
  __shared__ float tileMaxG[16][8];                // per-(row, wave) max, gold
  __shared__ float rowmaxS[2][16];
  __shared__ int   mrow[16];
  __shared__ int   lsum[16][16];
  __shared__ float resv[32];

  const int tid   = threadIdx.x;
  const int wave  = tid >> 5;        // 0..7 -> N-tile
  const int lane  = tid & 31;
  const int hi    = lane >> 4;       // lane half (K-run selector per ISA layout)
  const int n     = lane & 15;       // A row / D column within tile
  const int jbase = wave * 16;
  const int bbase = blockIdx.x * 16;

  // ---- expT B-tiles, register resident, shared by fwd+gold passes ----
  // B(KxN) wave32 layout: lane holds column jbase+n; K runs {8*hi..}, {16+8*hi..} per 32-chunk.
  v16h Bt[4];
#pragma unroll
  for (int c = 0; c < 4; ++c) {
    _Float16 tmp[16];
#pragma unroll
    for (int h = 0; h < 8; ++h) {
      int k0 = c*32 + 8*hi + h;
      int k1 = c*32 + 16 + 8*hi + h;
      float t0 = trans[k0*TT + jbase + n];
      float t1 = trans[k1*TT + jbase + n];
      tmp[h]     = (_Float16)((t0 == 0.0f) ? 0.0f : __expf(t0)); // trans==0 -> NEG -> exp 0
      tmp[h + 8] = (_Float16)((t1 == 0.0f) ? 0.0f : __expf(t1));
    }
    v16h v;
#pragma unroll
    for (int h = 0; h < 16; ++h) v[h] = tmp[h];
    Bt[c] = v;
  }

  // ---- t = 0 init ----
  {
    int row = tid >> 4, k0 = (tid & 15) * 8;
    long long base = (long long)(bbase + row) * SS * TT;
#pragma unroll
    for (int k = 0; k < 8; ++k) {
      int j = k0 + k;
      float a0 = startt[j] + feats[base + j];
      alphaF[row][j] = a0;
      alphaG[row][j] = ptags[base + j] ? a0 : NEGV;
    }
  }
  __syncthreads();
  // seed per-(row, wave) tile maxima for the first staging phase
  {
    int row = tid >> 4, seg = tid & 15;
    if (seg < 8) {
      float m = alphaF[row][seg * 16];
#pragma unroll
      for (int k = 1; k < 16; ++k) m = fmaxf(m, alphaF[row][seg * 16 + k]);
      tileMaxF[row][seg] = m;
    } else {
      int s2 = seg - 8;
      float m = alphaG[row][s2 * 16];
#pragma unroll
      for (int k = 1; k < 16; ++k) m = fmaxf(m, alphaG[row][s2 * 16 + k]);
      tileMaxG[row][s2] = m;
    }
  }
  __syncthreads();

  // ---- sequential scan, t = 1 .. S-1 : exactly 2 barriers per step ----
  for (int t = 1; t < SS; ++t) {
    const int row = tid >> 4, seg = tid & 15;

    // Phase B: rowmax from 8 per-wave partials, stage exp(alpha - m) as f16
    {
      float mF = tileMaxF[row][0], mG = tileMaxG[row][0];
#pragma unroll
      for (int w = 1; w < 8; ++w) {
        mF = fmaxf(mF, tileMaxF[row][w]);
        mG = fmaxf(mG, tileMaxG[row][w]);
      }
      if (seg == 0) { rowmaxS[0][row] = mF; rowmaxS[1][row] = mG; }
      if (tid < 16) mrow[tid] = maskI[(bbase + tid) * SS + t];

      long long pb = (long long)(bbase + row) * SS * TT + (long long)(t - 1) * TT + seg * 8;
#pragma unroll
      for (int k = 0; k < 8; ++k) {
        int j = seg * 8 + k;
        vhF[row][j] = (_Float16)__expf(alphaF[row][j] - mF);
        float eg = __expf(alphaG[row][j] - mG);
        vhG[row][j] = ptags[pb + k] ? (_Float16)eg : (_Float16)0.0f;  // prevmask = pt[t-1]
      }
    }
    __syncthreads();

    // Phase C: per-wave WMMA matvec-batch (16x128 @ 128x16), fwd then gold
    v8f dF = {}; v8f dG = {};
#pragma unroll
    for (int c = 0; c < 4; ++c) {
      v8h lo = *(const v8h*)&vhF[n][c*32 + 8*hi];
      v8h hh = *(const v8h*)&vhF[n][c*32 + 16 + 8*hi];
      v16h aF = cat8(lo, hh);
      dF = __builtin_amdgcn_wmma_f32_16x16x32_f16(false, aF, false, Bt[c],
                                                  (short)0, dF, false, false);
    }
#pragma unroll
    for (int c = 0; c < 4; ++c) {
      v8h lo = *(const v8h*)&vhG[n][c*32 + 8*hi];
      v8h hh = *(const v8h*)&vhG[n][c*32 + 16 + 8*hi];
      v16h aG = cat8(lo, hh);
      dG = __builtin_amdgcn_wmma_f32_16x16x32_f16(false, aG, false, Bt[c],
                                                  (short)0, dG, false, false);
    }

    // Update: new = m + log(S) + f ; select-by-mask (avoids -inf*0 NaN of blend form)
    // Then fold next step's row-max partials in via 16-lane butterflies (no barrier).
    {
      long long fb = (long long)t * TT + jbase + n;
#pragma unroll
      for (int r = 0; r < 8; ++r) {
        int b = r + 8*hi;
        long long gidx = (long long)(bbase + b) * SS * TT + fb;
        float f  = feats[gidx];
        int   mt = mrow[b];
        int   j  = jbase + n;

        float nf  = rowmaxS[0][b] + __logf(dF[r]) + f;
        float sFv = mt ? nf : alphaF[b][j];
        alphaF[b][j] = sFv;

        int   nm = ptags[gidx];                      // nextmask = pt[t]
        float sg = dG[r];
        float lg = (sg > 0.0f) ? __logf(sg) : -2.0e7f;
        float ng = nm ? fmaxf(rowmaxS[1][b] + lg + f, -3.0e7f) : -2.0e7f;
        float sGv = mt ? ng : alphaG[b][j];
        alphaG[b][j] = sGv;

        // per-row max over this wave's 16 columns (lanes of the same half)
        float rFm = sFv, rGm = sGv;
#pragma unroll
        for (int d2 = 1; d2 < 16; d2 <<= 1) {
          rFm = fmaxf(rFm, __shfl_xor(rFm, d2, 16));
          rGm = fmaxf(rGm, __shfl_xor(rGm, d2, 16));
        }
        if (n == 0) { tileMaxF[b][wave] = rFm; tileMaxG[b][wave] = rGm; }

        if (t + 1 < SS) __builtin_prefetch(&feats[gidx + TT], 0, 1); // next-step tile
      }
    }
    __syncthreads();
  }

  // ---- sequence lengths (parallel mask popcount) ----
  {
    int row = tid >> 4, seg = tid & 15;
    int base = (bbase + row) * SS + seg * 64;
    int c = 0;
    for (int s = 0; s < 64; ++s) c += maskI[base + s];
    lsum[row][seg] = c;
  }
  __syncthreads();

  // ---- terminal logsumexp: fwd (threads 0..15), gold (threads 16..31) ----
  if (tid < 32) {
    int isG = tid >> 4, b = tid & 15;
    int len = 0;
#pragma unroll
    for (int s = 0; s < 16; ++s) len += lsum[b][s];
    int last = len - 1;

    float m = -3.4e38f, acc = 0.0f;
    if (!isG) {
      for (int j = 0; j < TT; ++j) m = fmaxf(m, alphaF[b][j] + endt[j]);
      for (int j = 0; j < TT; ++j) acc += __expf(alphaF[b][j] + endt[j] - m);
    } else {
      long long pb = (long long)(bbase + b) * SS * TT + (long long)last * TT;
      for (int j = 0; j < TT; ++j) {
        float e = ptags[pb + j] ? ((endt[j] == 0.0f) ? NEGV : endt[j]) : NEGV;
        m = fmaxf(m, alphaG[b][j] + e);
      }
      for (int j = 0; j < TT; ++j) {
        float e = ptags[pb + j] ? ((endt[j] == 0.0f) ? NEGV : endt[j]) : NEGV;
        acc += __expf(alphaG[b][j] + e - m);
      }
    }
    resv[tid] = m + __logf(acc);
  }
  __syncthreads();
  if (tid == 0) {
    float s = 0.0f;
#pragma unroll
    for (int b = 0; b < 16; ++b) s += resv[b] - resv[16 + b];
    partial[blockIdx.x] = s;
  }
}

// Deterministic final reduction (no float atomics -> bitwise-stable output)
__global__ void crf_finalize_kernel(const float* __restrict__ partial,
                                    float* __restrict__ out) {
  out[0] = partial[0] + partial[1] + partial[2] + partial[3];
}

extern "C" void kernel_launch(void* const* d_in, const int* in_sizes, int n_in,
                              void* d_out, int out_size, void* d_ws, size_t ws_size,
                              hipStream_t stream) {
  const float* feats  = (const float*)d_in[0];
  // d_in[1] = tags: unused (reference only consumes possible_tags)
  const int*   maskI  = (const int*)  d_in[2];
  const int*   ptags  = (const int*)  d_in[3];
  const float* trans  = (const float*)d_in[4];
  const float* startt = (const float*)d_in[5];
  const float* endt   = (const float*)d_in[6];
  float* partial = (float*)d_ws;
  float* out     = (float*)d_out;

  crf_scan_kernel<<<dim3(4), dim3(256), 0, stream>>>(
      feats, maskI, ptags, trans, startt, endt, partial);
  crf_finalize_kernel<<<1, 1, 0, stream>>>(partial, out);
}
